// LlamaAttention_35759897707158
// MI455X (gfx1250) — matrix operand order repaired and checked by reading, not compiled, not run on hardware
//
#include <hip/hip_runtime.h>

// ---------------------------------------------------------------------------
// Types for WMMA fragments (gfx1250, wave32)
// ---------------------------------------------------------------------------
typedef __bf16 bf16x16 __attribute__((ext_vector_type(16)));
typedef __bf16 bf16x8  __attribute__((ext_vector_type(8)));
typedef __bf16 bf16x4  __attribute__((ext_vector_type(4)));
typedef float  f32x8   __attribute__((ext_vector_type(8)));

union FragBF16 { bf16x16 v; bf16x8 h[2]; };

#define T_TOK   4096
#define HIDDEN  2048
#define NQH     16
#define NKVH    4
#define HD      128
#define QKDIM   (NQH * HD)   // 2048
#define KVDIM   (NKVH * HD)  // 512

#if defined(__gfx1250__)
#define USE_ASYNC_LDS 1
#else
#define USE_ASYNC_LDS 0
#endif

// ---------------------------------------------------------------------------
// Generic bf16 WMMA GEMM:  C[M,N] = A[M,K] * B[K,N]
//   BM=128, BN=128, BK=32; 256 threads = 8 waves in a 4(m) x 2(n) grid;
//   each wave computes a 32x64 tile (2 A-frags, 4 B-frags, 8 WMMAs / K-step).
//   B tile is stored transposed [n][k] in LDS so B-fragments are contiguous.
// ---------------------------------------------------------------------------
template <bool A_IS_BF16, bool OUT_IS_F32>
__global__ __launch_bounds__(256) void gemm_bf16_wmma(
    const void* __restrict__ Aptr, const float* __restrict__ B,
    void* __restrict__ Cptr, int M, int N, int K, int lda, int ldb, int ldc)
{
    constexpr int BM = 128, BN = 128, BK = 32;
    __shared__ __bf16 lds_a[BM * BK];   // [m][k]   8 KB
    __shared__ __bf16 lds_b[BN * BK];   // [n][k]   8 KB (transposed tile)

    const int tid  = threadIdx.x;
    const int lane = tid & 31;
    const int wid  = tid >> 5;
    const int col  = lane & 15;
    const int half = lane >> 4;
    const int hoffA = half * 8;    // A fragment k-offset per ISA layout
    const int hoffB = half * 8;    // B fragment k-offset per ISA layout
    const int wm = (wid & 3) * 32; // wave row base inside block
    const int wn = (wid >> 2) * 64;// wave col base inside block
    const long m0 = (long)blockIdx.x * BM;
    const long n0 = (long)blockIdx.y * BN;

    f32x8 acc[2][4] = {};

    for (int k0 = 0; k0 < K; k0 += BK) {
        // ---- stage A tile ----
        if (!A_IS_BF16) {
            const float* A = (const float*)Aptr;
#pragma unroll
            for (int i = 0; i < 4; ++i) {             // 1024 chunks of 4 floats
                int c = tid + i * 256;
                int m = c >> 3, ks = (c & 7) * 4;
                float4 v = *(const float4*)&A[(m0 + m) * lda + k0 + ks];
                bf16x4 t; t[0] = (__bf16)v.x; t[1] = (__bf16)v.y;
                          t[2] = (__bf16)v.z; t[3] = (__bf16)v.w;
                *(bf16x4*)&lds_a[m * BK + ks] = t;
            }
        } else {
            const __bf16* A = (const __bf16*)Aptr;
#pragma unroll
            for (int i = 0; i < 2; ++i) {             // 512 chunks of 8 bf16
                int c = tid + i * 256;
                int m = c >> 2, ks = (c & 3) * 8;
                *(bf16x8*)&lds_a[m * BK + ks] =
                    *(const bf16x8*)&A[(m0 + m) * lda + k0 + ks];
            }
        }
        // ---- stage B tile (transposed into [n][k]) ----
#pragma unroll
        for (int i = 0; i < 4; ++i) {                 // 1024 chunks of 4 floats
            int c = tid + i * 256;
            int k = c >> 5, ns = (c & 31) * 4;
            float4 v = *(const float4*)&B[(long)(k0 + k) * ldb + n0 + ns];
            lds_b[(ns + 0) * BK + k] = (__bf16)v.x;
            lds_b[(ns + 1) * BK + k] = (__bf16)v.y;
            lds_b[(ns + 2) * BK + k] = (__bf16)v.z;
            lds_b[(ns + 3) * BK + k] = (__bf16)v.w;
        }
        __syncthreads();

        // ---- WMMA compute: wave tile 32 x 64 ----
        FragBF16 af[2];
#pragma unroll
        for (int mt = 0; mt < 2; ++mt) {
            const __bf16* ap = &lds_a[(wm + mt * 16 + col) * BK + hoffA];
            af[mt].h[0] = *(const bf16x8*)(ap);
            af[mt].h[1] = *(const bf16x8*)(ap + 16);
        }
#pragma unroll
        for (int nt = 0; nt < 4; ++nt) {
            FragBF16 bfr;
            const __bf16* bp = &lds_b[(wn + nt * 16 + col) * BK + hoffB];
            bfr.h[0] = *(const bf16x8*)(bp);
            bfr.h[1] = *(const bf16x8*)(bp + 16);
#pragma unroll
            for (int mt = 0; mt < 2; ++mt)
                acc[mt][nt] = __builtin_amdgcn_wmma_f32_16x16x32_bf16(
                    false, af[mt].v, false, bfr.v, (short)0, acc[mt][nt],
                    false, false);
        }
        __syncthreads();
    }

    // ---- store (C layout: row = g + 8*half, col = lane&15) ----
#pragma unroll
    for (int mt = 0; mt < 2; ++mt)
#pragma unroll
        for (int nt = 0; nt < 4; ++nt)
#pragma unroll
            for (int g = 0; g < 8; ++g) {
                long m = m0 + wm + mt * 16 + g + 8 * half;
                long n = n0 + wn + nt * 16 + col;
                if (OUT_IS_F32)
                    ((float*)Cptr)[m * ldc + n] = acc[mt][nt][g];
                else
                    ((__bf16*)Cptr)[m * ldc + n] = (__bf16)acc[mt][nt][g];
            }
}

// ---------------------------------------------------------------------------
// RoPE (in-place on bf16 Q and K), pos/seq derived from cu_seqlens.
// ---------------------------------------------------------------------------
__global__ __launch_bounds__(256) void rope_kernel(
    __bf16* __restrict__ Q, __bf16* __restrict__ Kb, const int* __restrict__ cu)
{
    int idx = blockIdx.x * blockDim.x + threadIdx.x;
    int d    = idx & 63;
    int rest = idx >> 6;
    int hh   = rest % (NQH + NKVH);
    int t    = rest / (NQH + NKVH);
    if (t >= T_TOK) return;

    int c1 = cu[1], c2 = cu[2], c3 = cu[3];
    int s = 0; if (t >= c1) s = 1; if (t >= c2) s = 2; if (t >= c3) s = 3;
    int pos = t - cu[s];

    // inv_freq = 10000^(-d/64) = exp(-d * ln(10000)/64)
    float f = (float)pos * __expf(-(float)d * 0.14391156514261f);
    float cs = __cosf(f), sn = __sinf(f);

    __bf16* base = (hh < NQH) ? (Q  + (long)t * QKDIM + hh * HD)
                              : (Kb + (long)t * KVDIM + (hh - NQH) * HD);
    float x1 = (float)base[d], x2 = (float)base[d + 64];
    base[d]      = (__bf16)(x1 * cs - x2 * sn);
    base[d + 64] = (__bf16)(x2 * cs + x1 * sn);
}

// ---------------------------------------------------------------------------
// Flash attention, block-diagonal causal (packed sequences).
// Grid: (T/64, NQH). Block: 128 threads = 4 waves; wave w owns q-rows
// [qt*64 + w*16, +16). Q stays in registers as 4 A-fragments (HD=128).
// K tile streamed global->LDS with GLOBAL_LOAD_ASYNC_TO_LDS_B128 (ASYNCcnt);
// V tile transposed [hd][key] via registers; P via per-wave LDS roundtrip.
// ---------------------------------------------------------------------------
__global__ __launch_bounds__(128) void attn_kernel(
    const __bf16* __restrict__ Q, const __bf16* __restrict__ Kb,
    const __bf16* __restrict__ Vb, const int* __restrict__ cu,
    __bf16* __restrict__ Ob)
{
    __shared__ __bf16 lds_k[64 * 128];      // [key][hd]   16 KB
    __shared__ __bf16 lds_vt[128 * 64];     // [hd][key]   16 KB
    __shared__ __bf16 lds_p[4 * 16 * 64];   // per-wave P   8 KB

    const int tid  = threadIdx.x;
    const int lane = tid & 31;
    const int wid  = tid >> 5;
    const int col  = lane & 15;
    const int half = lane >> 4;
    const int hoffA = half * 8;
    const int hoffB = half * 8;

    const int qt  = blockIdx.x;
    const int h   = blockIdx.y;
    const int kvh = h >> 2;                  // GQA: 4 q-heads per kv-head

    const int c1 = cu[1], c2 = cu[2], c3 = cu[3];
    const int qrow_base = qt * 64 + wid * 16;

    // Q fragments (registers), row = lane&15 of the wave stripe
    FragBF16 qf[4];
    {
        const __bf16* qp = Q + (long)(qrow_base + col) * QKDIM + h * HD;
#pragma unroll
        for (int f = 0; f < 4; ++f) {
            qf[f].h[0] = *(const bf16x8*)(qp + f * 32 + hoffA);
            qf[f].h[1] = *(const bf16x8*)(qp + f * 32 + hoffA + 16);
        }
    }

    int seqq[8];
#pragma unroll
    for (int g = 0; g < 8; ++g) {
        int t = qrow_base + g + 8 * half;
        int s = 0; if (t >= c1) s = 1; if (t >= c2) s = 2; if (t >= c3) s = 3;
        seqq[g] = s;
    }

    f32x8 o[8] = {};
    float rowM[8], rowL[8];
#pragma unroll
    for (int g = 0; g < 8; ++g) { rowM[g] = -1e30f; rowL[g] = 0.f; }

    // block-diagonal skip: keys before this q-block's sequence start are masked
    int qb = qt * 64;
    int sb = 0; if (qb >= c1) sb = 1; if (qb >= c2) sb = 2; if (qb >= c3) sb = 3;
    int kt_start = cu[sb] >> 6;

    const float scale = 0.08838834764831845f;  // 1/sqrt(128)

    for (int kt = kt_start; kt <= qt; ++kt) {
        const int kbase = kt * 64;
        // ---- stage K tile [key][hd] (async DMA) + V tile transposed ----
#pragma unroll
        for (int i = 0; i < 8; ++i) {          // 1024 chunks of 8 bf16
            int c   = tid + i * 128;
            int key = c >> 4, seg = c & 15;
            long roff = (long)(kbase + key) * KVDIM + kvh * HD + seg * 8;
            __bf16* lk = &lds_k[key * 128 + seg * 8];
#if USE_ASYNC_LDS
            const __bf16* gk = &Kb[roff];
            unsigned lds_off = (unsigned)(unsigned long long)lk;
            asm volatile("global_load_async_to_lds_b128 %0, %1, off"
                         :: "v"(lds_off), "v"(gk) : "memory");
#else
            *(bf16x8*)lk = *(const bf16x8*)&Kb[roff];
#endif
            bf16x8 vv = *(const bf16x8*)&Vb[roff];
#pragma unroll
            for (int j = 0; j < 8; ++j) lds_vt[(seg * 8 + j) * 64 + key] = vv[j];
        }
#if USE_ASYNC_LDS
        asm volatile("s_wait_asynccnt 0x0" ::: "memory");
#endif
        __syncthreads();

        // ---- S = Q * K^T  (16 x 64 stripe per wave) ----
        f32x8 s[4] = {};
#pragma unroll
        for (int nt = 0; nt < 4; ++nt)
#pragma unroll
            for (int f = 0; f < 4; ++f) {
                FragBF16 bfr;
                const __bf16* kp = &lds_k[(nt * 16 + col) * 128 + f * 32 + hoffB];
                bfr.h[0] = *(const bf16x8*)(kp);
                bfr.h[1] = *(const bf16x8*)(kp + 16);
                s[nt] = __builtin_amdgcn_wmma_f32_16x16x32_bf16(
                    false, qf[f].v, false, bfr.v, (short)0, s[nt], false, false);
            }

        // ---- scale + block-diagonal causal mask ----
#pragma unroll
        for (int nt = 0; nt < 4; ++nt) {
            int kidx = kbase + nt * 16 + col;
            int sk = 0; if (kidx >= c1) sk = 1; if (kidx >= c2) sk = 2; if (kidx >= c3) sk = 3;
#pragma unroll
            for (int g = 0; g < 8; ++g) {
                int qidx = qrow_base + g + 8 * half;
                bool ok = (sk == seqq[g]) && (kidx <= qidx);
                s[nt][g] = ok ? s[nt][g] * scale : -1e30f;
            }
        }

        // ---- online softmax (row reduce across 16 lanes of the half) ----
        float alpha[8];
#pragma unroll
        for (int g = 0; g < 8; ++g) {
            float m = fmaxf(fmaxf(s[0][g], s[1][g]), fmaxf(s[2][g], s[3][g]));
#pragma unroll
            for (int d = 1; d < 16; d <<= 1) m = fmaxf(m, __shfl_xor(m, d, 32));
            float Mn = fmaxf(rowM[g], m);
            alpha[g] = __expf(rowM[g] - Mn);
            rowM[g]  = Mn;
            float lsum = 0.f;
#pragma unroll
            for (int nt = 0; nt < 4; ++nt) {
                float p = __expf(s[nt][g] - Mn);
                if (s[nt][g] <= -1e29f) p = 0.f;   // fully-masked guard
                s[nt][g] = p;
                lsum += p;
            }
#pragma unroll
            for (int d = 1; d < 16; d <<= 1) lsum += __shfl_xor(lsum, d, 32);
            rowL[g] = rowL[g] * alpha[g] + lsum;
        }
#pragma unroll
        for (int t = 0; t < 8; ++t)
#pragma unroll
            for (int g = 0; g < 8; ++g) o[t][g] *= alpha[g];

        // ---- P: C-layout -> A-layout via per-wave LDS region ----
        __bf16* pw = &lds_p[wid * 16 * 64];
#pragma unroll
        for (int nt = 0; nt < 4; ++nt)
#pragma unroll
            for (int g = 0; g < 8; ++g)
                pw[(g + 8 * half) * 64 + nt * 16 + col] = (__bf16)s[nt][g];
        __syncthreads();

        // ---- O += P * V ----
#pragma unroll
        for (int f = 0; f < 2; ++f) {
            FragBF16 pf;
            pf.h[0] = *(const bf16x8*)&pw[col * 64 + f * 32 + hoffA];
            pf.h[1] = *(const bf16x8*)&pw[col * 64 + f * 32 + hoffA + 16];
#pragma unroll
            for (int nt = 0; nt < 8; ++nt) {
                FragBF16 vf;
                const __bf16* vp = &lds_vt[(nt * 16 + col) * 64 + f * 32 + hoffB];
                vf.h[0] = *(const bf16x8*)(vp);
                vf.h[1] = *(const bf16x8*)(vp + 16);
                o[nt] = __builtin_amdgcn_wmma_f32_16x16x32_bf16(
                    false, pf.v, false, vf.v, (short)0, o[nt], false, false);
            }
        }
        __syncthreads();
    }

    // ---- normalize + store attention output (bf16, [T, H*HD]) ----
#pragma unroll
    for (int g = 0; g < 8; ++g) {
        float inv = 1.f / rowL[g];
        long m = qrow_base + g + 8 * half;
#pragma unroll
        for (int nt = 0; nt < 8; ++nt)
            Ob[m * QKDIM + h * HD + nt * 16 + col] = (__bf16)(o[nt][g] * inv);
    }
}

// ---------------------------------------------------------------------------
// Launcher
// ---------------------------------------------------------------------------
extern "C" void kernel_launch(void* const* d_in, const int* in_sizes, int n_in,
                              void* d_out, int out_size, void* d_ws, size_t ws_size,
                              hipStream_t stream)
{
    (void)in_sizes; (void)n_in; (void)out_size; (void)ws_size;
    const float* hidden = (const float*)d_in[0];
    const float* Wq     = (const float*)d_in[1];
    const float* Wk     = (const float*)d_in[2];
    const float* Wv     = (const float*)d_in[3];
    const float* Wo     = (const float*)d_in[4];
    const int*   cu     = (const int*)d_in[5];
    float*       out    = (float*)d_out;

    // workspace: Q(16MB) | K(4MB) | V(4MB) | attn-out(16MB), all bf16
    char* ws = (char*)d_ws;
    __bf16* Qb = (__bf16*)ws;
    __bf16* Kb = (__bf16*)(ws + (size_t)T_TOK * QKDIM * 2);
    __bf16* Vb = (__bf16*)(ws + (size_t)T_TOK * QKDIM * 2 + (size_t)T_TOK * KVDIM * 2);
    __bf16* Ab = (__bf16*)(ws + (size_t)T_TOK * QKDIM * 2 + 2 * (size_t)T_TOK * KVDIM * 2);

    dim3 blk(256);
    // QKV projections (f32 in -> bf16 out)
    gemm_bf16_wmma<false, false><<<dim3(T_TOK / 128, QKDIM / 128), blk, 0, stream>>>(
        hidden, Wq, Qb, T_TOK, QKDIM, HIDDEN, HIDDEN, QKDIM, QKDIM);
    gemm_bf16_wmma<false, false><<<dim3(T_TOK / 128, KVDIM / 128), blk, 0, stream>>>(
        hidden, Wk, Kb, T_TOK, KVDIM, HIDDEN, HIDDEN, KVDIM, KVDIM);
    gemm_bf16_wmma<false, false><<<dim3(T_TOK / 128, KVDIM / 128), blk, 0, stream>>>(
        hidden, Wv, Vb, T_TOK, KVDIM, HIDDEN, HIDDEN, KVDIM, KVDIM);

    // RoPE on Q and K
    {
        int total = T_TOK * (NQH + NKVH) * 64;
        rope_kernel<<<(total + 255) / 256, 256, 0, stream>>>(Qb, Kb, cu);
    }

    // Flash attention, block-diagonal causal
    attn_kernel<<<dim3(T_TOK / 64, NQH), dim3(128), 0, stream>>>(Qb, Kb, Vb, cu, Ab);

    // Output projection (bf16 A, f32 out)
    gemm_bf16_wmma<true, true><<<dim3(T_TOK / 128, HIDDEN / 128), blk, 0, stream>>>(
        Ab, Wo, out, T_TOK, HIDDEN, QKDIM, QKDIM, HIDDEN, HIDDEN);
}
